// MolecularGNN_61521111547945
// MI455X (gfx1250) — matrix operand order repaired and checked by reading, not compiled, not run on hardware
//
#include <hip/hip_runtime.h>
#include <hip/hip_bf16.h>
#include <stdint.h>

#define HID   128
#define HEADS 8
#define CDIM  16
#define LAYERS 4
#define EPSV 1e-5f

typedef __attribute__((ext_vector_type(16))) _Float16 v16h;
typedef __attribute__((ext_vector_type(8)))  _Float16 v8h;
typedef __attribute__((ext_vector_type(8)))  float    v8f;

// ---------- helpers ----------
__device__ __forceinline__ void atomicMaxF(float* addr, float val) {
  // monotonic-in-bits trick: correct for all non-NaN floats incl. +/-inf
  if (val >= 0.f) atomicMax((int*)addr, __float_as_int(val));
  else            atomicMin((unsigned int*)addr, __float_as_uint(val));
}

__global__ void k_fill_f32(float* __restrict__ p, long n, float v) {
  long i = (long)blockIdx.x * blockDim.x + threadIdx.x;
  if (i < n) p[i] = v;
}

// ---------- weight preprocessing ----------
// Wt[n,k] = (f16) W[k,n]   (row-major [K,Nout] -> row-major [Nout,K])
__global__ void k_transpose_f16(const float* __restrict__ W, _Float16* __restrict__ Wt,
                                int K, int Nout) {
  long i = (long)blockIdx.x * blockDim.x + threadIdx.x;
  if (i >= (long)K * Nout) return;
  int k = (int)(i / Nout), n = (int)(i % Nout);
  Wt[(size_t)n * K + k] = (_Float16)W[i];
}

// w_edge_att[l][k][h] = sum_c lin_edge_W[l][k][h*16+c] * att_edge[l][h][c]
__global__ void k_wea(const float* __restrict__ linEW, const float* __restrict__ attE,
                      float* __restrict__ wea) {
  int i = blockIdx.x * blockDim.x + threadIdx.x;
  if (i >= LAYERS * HID * HEADS) return;
  int l = i / (HID * HEADS);
  int r = i % (HID * HEADS);
  int k = r / HEADS, hh = r % HEADS;
  float s = 0.f;
  const float* wrow = linEW + ((size_t)l * HID + k) * HID + hh * CDIM;
  const float* av   = attE + ((size_t)l * HEADS + hh) * CDIM;
  #pragma unroll
  for (int c = 0; c < CDIM; c++) s += wrow[c] * av[c];
  wea[i] = s;
}

// ---------- embeddings ----------
__global__ void k_node_embed(const float* __restrict__ x, const float* __restrict__ W,
                             const float* __restrict__ b, _Float16* __restrict__ h16,
                             float* __restrict__ residual, long total) {
  long i = (long)blockIdx.x * blockDim.x + threadIdx.x;
  if (i >= total) return;
  long n = i >> 7; int c = (int)(i & 127);
  float acc = b[c];
  const float* xr = x + n * 9;
  #pragma unroll
  for (int k = 0; k < 9; k++) acc += xr[k] * W[k * HID + c];
  h16[i] = (_Float16)acc;
  residual[i] = acc;
}

__global__ void k_edge_embed(const float* __restrict__ ea, const float* __restrict__ W,
                             const float* __restrict__ b, _Float16* __restrict__ ea16,
                             long total) {
  long i = (long)blockIdx.x * blockDim.x + threadIdx.x;
  if (i >= total) return;
  long e = i >> 7; int c = (int)(i & 127);
  const float* er = ea + e * 3;
  float acc = b[c] + er[0] * W[0 * HID + c] + er[1] * W[1 * HID + c] + er[2] * W[2 * HID + c];
  ea16[i] = (_Float16)acc;
}

// ---------- WMMA GEMM: out[M,Nout] = A[M,K](f16,row) @ Wt[Nout,K](f16,row)^T ----------
// one wave computes one 16x16 output tile; K multiple of 32, M/Nout multiples of 16
__global__ void k_wmma_gemm(const _Float16* __restrict__ A, const _Float16* __restrict__ Wt,
                            const float* __restrict__ bias,
                            float* __restrict__ outF, _Float16* __restrict__ outH,
                            int M, int K, int Nout, int doRelu) {
  int ntiles = Nout >> 4;
  long wave  = (long)blockIdx.x * (blockDim.x >> 5) + (threadIdx.x >> 5);
  long total = (long)(M >> 4) * ntiles;
  if (wave >= total) return;               // wave-uniform: EXEC all-1 inside WMMA
  int tm = (int)(wave / ntiles);
  int tn = (int)(wave % ntiles);
  int lane  = threadIdx.x & 31;
  int halfw = lane >> 4;                   // lane half selects K sub-block
  int mr    = lane & 15;

  const _Float16* arow = A  + (size_t)(tm * 16 + mr) * K;   // A row m
  const _Float16* brow = Wt + (size_t)(tn * 16 + mr) * K;   // B column n (transposed weights)

  v8f c = {};
  for (int kk = 0; kk < K; kk += 32) {
    // A 16-bit layout: halves 0-7 -> k = kk + halfw*8 + i ; halves 8-15 -> +16
    v8h lo = *(const v8h*)(arow + kk + halfw * 8);
    v8h hi = *(const v8h*)(arow + kk + halfw * 8 + 16);
    v16h a;
    #pragma unroll
    for (int i = 0; i < 8; i++) { a[i] = lo[i]; a[i + 8] = hi[i]; }
    // B 16-bit layout: halves 0-7 -> k = kk + halfw*8 + i ; halves 8-15 -> +16, column n = mr
    v8h blo = *(const v8h*)(brow + kk + halfw * 8);
    v8h bhi = *(const v8h*)(brow + kk + halfw * 8 + 16);
    v16h b;
    #pragma unroll
    for (int i = 0; i < 8; i++) { b[i] = blo[i]; b[i + 8] = bhi[i]; }
    c = __builtin_amdgcn_wmma_f32_16x16x32_f16(false, a, false, b, (short)0, c, false, false);
  }

  int col = tn * 16 + mr;
  float bv = bias ? bias[col] : 0.f;
  #pragma unroll
  for (int r = 0; r < 8; r++) {
    int row = tm * 16 + r + halfw * 8;     // C/D layout: vgpr r -> m = r + 8*half
    float v = c[r] + bv;
    if (doRelu) v = fmaxf(v, 0.f);
    size_t o = (size_t)row * Nout + col;
    if (outF) outF[o] = v;
    if (outH) outH[o] = (_Float16)v;
  }
}

// ---------- per-layer attention ----------
__global__ void k_alpha_node(const float* __restrict__ hs, const float* __restrict__ attS,
                             const float* __restrict__ attD, float* __restrict__ aS,
                             float* __restrict__ aD, int N) {
  int n = blockIdx.x * blockDim.x + threadIdx.x;
  if (n >= N) return;
  const float* hr = hs + (size_t)n * HID;
  #pragma unroll
  for (int h = 0; h < HEADS; h++) {
    float s1 = 0.f, s2 = 0.f;
    #pragma unroll
    for (int c = 0; c < CDIM; c++) {
      float v = hr[h * CDIM + c];
      s1 += v * attS[h * CDIM + c];
      s2 += v * attD[h * CDIM + c];
    }
    aS[(size_t)n * HEADS + h] = s1;
    aD[(size_t)n * HEADS + h] = s2;
  }
}

__global__ void k_init_out(float* __restrict__ out, const float* __restrict__ gatb, long total) {
  long i = (long)blockIdx.x * blockDim.x + threadIdx.x;
  if (i < total) out[i] = gatb[i & 127];
}

__global__ void k_edge_logits(const _Float16* __restrict__ ea16, const float* __restrict__ wea,
                              const float* __restrict__ aS, const float* __restrict__ aD,
                              const int* __restrict__ srcI, const int* __restrict__ dstI,
                              float* __restrict__ aE, float* __restrict__ mbuf, long EH) {
  long t = (long)blockIdx.x * blockDim.x + threadIdx.x;
  if (t >= EH) return;
  long e = t >> 3; int hh = (int)(t & 7);
  int sN = srcI[e], dN = dstI[e];
  const _Float16* er = ea16 + (size_t)e * HID;
  float dotv = 0.f;
  #pragma unroll 8
  for (int c = 0; c < HID; c++) dotv += (float)er[c] * wea[c * HEADS + hh];
  float l = aS[(size_t)sN * HEADS + hh] + aD[(size_t)dN * HEADS + hh] + dotv;
  l = (l > 0.f) ? l : 0.2f * l;            // leaky relu 0.2
  aE[t] = l;
  atomicMaxF(&mbuf[(size_t)dN * HEADS + hh], l);
}

__global__ void k_edge_exp(const int* __restrict__ dstI, const float* __restrict__ mbuf,
                           float* __restrict__ aE, float* __restrict__ sbuf, long EH) {
  long t = (long)blockIdx.x * blockDim.x + threadIdx.x;
  if (t >= EH) return;
  long e = t >> 3; int hh = (int)(t & 7);
  int dN = dstI[e];
  float v = expf(aE[t] - mbuf[(size_t)dN * HEADS + hh]);
  aE[t] = v;
  atomicAdd(&sbuf[(size_t)dN * HEADS + hh], v);
}

__global__ void k_scatter(const float* __restrict__ hs, const float* __restrict__ aE,
                          const float* __restrict__ sbuf, const int* __restrict__ srcI,
                          const int* __restrict__ dstI, float* __restrict__ out, long EH) {
  long t = (long)blockIdx.x * blockDim.x + threadIdx.x;
  if (t >= EH) return;
  long e = t >> 3; int hh = (int)(t & 7);
  int sN = srcI[e], dN = dstI[e];
  float w = aE[t] / sbuf[(size_t)dN * HEADS + hh];
  const float* hrow = hs + (size_t)sN * HID + hh * CDIM;
  float* orow = out + (size_t)dN * HID + hh * CDIM;
  #pragma unroll
  for (int c = 0; c < CDIM; c++) atomicAdd(&orow[c], hrow[c] * w);
}

// ---------- batchnorm ----------
__global__ void k_bn_stats(const float* __restrict__ out, float* __restrict__ bnsum,
                           float* __restrict__ bnsq, int rowsPerBlock) {
  int c = threadIdx.x;                       // blockDim.x == 128
  long r0 = (long)blockIdx.x * rowsPerBlock;
  float s = 0.f, s2 = 0.f;
  for (int r = 0; r < rowsPerBlock; r++) {
    float v = out[(r0 + r) * HID + c];
    s += v; s2 += v * v;
  }
  atomicAdd(&bnsum[c], s);
  atomicAdd(&bnsq[c], s2);
}

__global__ void k_bn_norm(const float* __restrict__ out, const float* __restrict__ bnsum,
                          const float* __restrict__ bnsq, const float* __restrict__ gamma,
                          const float* __restrict__ beta, float* __restrict__ residual,
                          _Float16* __restrict__ h16, int addRes, long total, float invN) {
  long i = (long)blockIdx.x * blockDim.x + threadIdx.x;
  if (i >= total) return;
  int c = (int)(i & 127);
  float mu  = bnsum[c] * invN;
  float var = bnsq[c] * invN - mu * mu;
  float v = (out[i] - mu) * rsqrtf(var + EPSV) * gamma[c] + beta[c];
  v = fmaxf(v, 0.f);
  if (addRes) { v += residual[i]; residual[i] = v; }
  h16[i] = (_Float16)v;
}

// ---------- pooling ----------
__global__ void k_pool_acc(const _Float16* __restrict__ h16, const int* __restrict__ batch,
                           float* __restrict__ cnt, float* __restrict__ psum,
                           float* __restrict__ pmax, long total) {
  long i = (long)blockIdx.x * blockDim.x + threadIdx.x;
  if (i >= total) return;
  long n = i >> 7; int c = (int)(i & 127);
  int b = batch[n];
  float v = (float)h16[i];
  atomicAdd(&psum[(size_t)b * HID + c], v);
  atomicMaxF(&pmax[(size_t)b * HID + c], v);
  if (c == 0) atomicAdd(&cnt[b], 1.f);
}

__global__ void k_gate(const _Float16* __restrict__ h16, const float* __restrict__ g1W,
                       const float* __restrict__ g1b, const float* __restrict__ g2W,
                       const float* __restrict__ g2b, const int* __restrict__ batch,
                       float* __restrict__ gate, float* __restrict__ gmax, int N) {
  __shared__ float W1[HID * 64];
  __shared__ float b1[64];
  __shared__ float W2[64];
  for (int idx = threadIdx.x; idx < HID * 64; idx += blockDim.x) W1[idx] = g1W[idx];
  if (threadIdx.x < 64) { b1[threadIdx.x] = g1b[threadIdx.x]; W2[threadIdx.x] = g2W[threadIdx.x]; }
  __syncthreads();
  int n = blockIdx.x * blockDim.x + threadIdx.x;
  if (n >= N) return;
  const _Float16* hr = h16 + (size_t)n * HID;
  float acc = g2b[0];
  for (int j = 0; j < 64; j++) {
    float t = b1[j];
    #pragma unroll 8
    for (int c = 0; c < HID; c++) t += (float)hr[c] * W1[c * 64 + j];
    acc += fmaxf(t, 0.f) * W2[j];
  }
  gate[n] = acc;
  atomicMaxF(&gmax[batch[n]], acc);
}

__global__ void k_gate_exp(const int* __restrict__ batch, const float* __restrict__ gmax,
                           float* __restrict__ gate, float* __restrict__ gsum, int N) {
  int n = blockIdx.x * blockDim.x + threadIdx.x;
  if (n >= N) return;
  float e = expf(gate[n] - gmax[batch[n]]);
  gate[n] = e;
  atomicAdd(&gsum[batch[n]], e);
}

__global__ void k_att_pool(const _Float16* __restrict__ h16, const int* __restrict__ batch,
                           const float* __restrict__ gate, const float* __restrict__ gsum,
                           float* __restrict__ patt, long total) {
  long i = (long)blockIdx.x * blockDim.x + threadIdx.x;
  if (i >= total) return;
  long n = i >> 7; int c = (int)(i & 127);
  int b = batch[n];
  atomicAdd(&patt[(size_t)b * HID + c], (gate[n] / gsum[b]) * (float)h16[i]);
}

__global__ void k_assemble(const float* __restrict__ psum, const float* __restrict__ pmax,
                           const float* __restrict__ patt, const float* __restrict__ cnt,
                           _Float16* __restrict__ g16, long total) {
  long i = (long)blockIdx.x * blockDim.x + threadIdx.x;
  if (i >= total) return;
  long b = i / 384; int j = (int)(i % 384);
  float v;
  if (j < 128)       v = psum[b * HID + j] / fmaxf(cnt[b], 1.f);
  else if (j < 256) { float pm = pmax[b * HID + (j - 128)]; v = (pm > -1e38f) ? pm : 0.f; }
  else               v = patt[b * HID + (j - 256)];
  g16[i] = (_Float16)v;
}

// ---------- launch ----------
extern "C" void kernel_launch(void* const* d_in, const int* in_sizes, int n_in,
                              void* d_out, int out_size, void* d_ws, size_t ws_size,
                              hipStream_t stream) {
  const float* x        = (const float*)d_in[0];
  const float* edge_attr= (const float*)d_in[1];
  const float* node_W   = (const float*)d_in[2];
  const float* node_b   = (const float*)d_in[3];
  const float* edge_W   = (const float*)d_in[4];
  const float* edge_b   = (const float*)d_in[5];
  const float* lin_W    = (const float*)d_in[6];
  const float* lin_eW   = (const float*)d_in[7];
  const float* att_src  = (const float*)d_in[8];
  const float* att_dst  = (const float*)d_in[9];
  const float* att_edge = (const float*)d_in[10];
  const float* gat_b    = (const float*)d_in[11];
  const float* bn_gamma = (const float*)d_in[12];
  const float* bn_beta  = (const float*)d_in[13];
  const float* g1W      = (const float*)d_in[14];
  const float* g1b      = (const float*)d_in[15];
  const float* g2W      = (const float*)d_in[16];
  const float* g2b      = (const float*)d_in[17];
  const float* mlp1_W   = (const float*)d_in[18];
  const float* mlp1_b   = (const float*)d_in[19];
  const float* mlp2_W   = (const float*)d_in[20];
  const float* mlp2_b   = (const float*)d_in[21];
  const int*   edge_idx = (const int*)d_in[22];
  const int*   batch    = (const int*)d_in[23];

  const long N = in_sizes[0] / 9;
  const long E = in_sizes[1] / 3;
  const long B = out_size / HID;
  const int* srcI = edge_idx;
  const int* dstI = edge_idx + E;

  // workspace bump allocator (256B aligned)
  char* wp = (char*)d_ws;
  auto alloc = [&](size_t bytes) -> void* {
    void* p = (void*)wp; wp += (bytes + 255) & ~(size_t)255; return p;
  };
  float*    residual = (float*)   alloc(N * HID * 4);
  float*    hs       = (float*)   alloc(N * HID * 4);
  float*    outb     = (float*)   alloc(N * HID * 4);
  _Float16* h16      = (_Float16*)alloc(N * HID * 2);
  _Float16* ea16     = (_Float16*)alloc(E * HID * 2);
  float*    aE       = (float*)   alloc(E * HEADS * 4);
  float*    aS       = (float*)   alloc(N * HEADS * 4);
  float*    aD       = (float*)   alloc(N * HEADS * 4);
  float*    mbuf     = (float*)   alloc(N * HEADS * 4);
  float*    sbuf     = (float*)   alloc(N * HEADS * 4);
  float*    bnsum    = (float*)   alloc(HID * 4);
  float*    bnsq     = (float*)   alloc(HID * 4);
  float*    wea      = (float*)   alloc(LAYERS * HID * HEADS * 4);
  _Float16* linWt    = (_Float16*)alloc((size_t)LAYERS * HID * HID * 2);
  _Float16* m1Wt     = (_Float16*)alloc((size_t)256 * 384 * 2);
  _Float16* m2Wt     = (_Float16*)alloc((size_t)128 * 256 * 2);
  float*    gateb    = (float*)   alloc(N * 4);
  float*    cnt      = (float*)   alloc(B * 4);
  float*    psum     = (float*)   alloc(B * HID * 4);
  float*    pmax     = (float*)   alloc(B * HID * 4);
  float*    patt     = (float*)   alloc(B * HID * 4);
  float*    gmax     = (float*)   alloc(B * 4);
  float*    gsum     = (float*)   alloc(B * 4);
  _Float16* g16      = (_Float16*)alloc(B * 384 * 2);
  _Float16* t1_16    = (_Float16*)alloc(B * 256 * 2);
  (void)ws_size; (void)n_in;

  const int TB = 256;
  auto blocks = [](long n, int tb) { return (unsigned)((n + tb - 1) / tb); };
  auto fill = [&](float* p, long n, float v) {
    k_fill_f32<<<blocks(n, TB), TB, 0, stream>>>(p, n, v);
  };

  // --- weight prep ---
  for (int l = 0; l < LAYERS; l++)
    k_transpose_f16<<<blocks(HID * HID, TB), TB, 0, stream>>>(
        lin_W + (size_t)l * HID * HID, linWt + (size_t)l * HID * HID, HID, HID);
  k_transpose_f16<<<blocks(384 * 256, TB), TB, 0, stream>>>(mlp1_W, m1Wt, 384, 256);
  k_transpose_f16<<<blocks(256 * 128, TB), TB, 0, stream>>>(mlp2_W, m2Wt, 256, 128);
  k_wea<<<blocks(LAYERS * HID * HEADS, TB), TB, 0, stream>>>(lin_eW, att_edge, wea);

  // --- embeddings ---
  k_node_embed<<<blocks(N * HID, TB), TB, 0, stream>>>(x, node_W, node_b, h16, residual, N * HID);
  k_edge_embed<<<blocks(E * HID, TB), TB, 0, stream>>>(edge_attr, edge_W, edge_b, ea16, E * HID);

  // --- GAT layers ---
  const long EH = E * HEADS;
  for (int l = 0; l < LAYERS; l++) {
    long waves = (N >> 4) * (HID >> 4);
    k_wmma_gemm<<<blocks(waves * 32, TB), TB, 0, stream>>>(
        h16, linWt + (size_t)l * HID * HID, nullptr, hs, nullptr,
        (int)N, HID, HID, 0);
    k_alpha_node<<<blocks(N, TB), TB, 0, stream>>>(
        hs, att_src + (size_t)l * HID, att_dst + (size_t)l * HID, aS, aD, (int)N);
    fill(mbuf, N * HEADS, -INFINITY);
    fill(sbuf, N * HEADS, 0.f);
    fill(bnsum, HID, 0.f);
    fill(bnsq, HID, 0.f);
    k_init_out<<<blocks(N * HID, TB), TB, 0, stream>>>(outb, gat_b + (size_t)l * HID, N * HID);
    k_edge_logits<<<blocks(EH, TB), TB, 0, stream>>>(
        ea16, wea + (size_t)l * HID * HEADS, aS, aD, srcI, dstI, aE, mbuf, EH);
    k_edge_exp<<<blocks(EH, TB), TB, 0, stream>>>(dstI, mbuf, aE, sbuf, EH);
    k_scatter<<<blocks(EH, TB), TB, 0, stream>>>(hs, aE, sbuf, srcI, dstI, outb, EH);
    k_bn_stats<<<(unsigned)(N / 256), 128, 0, stream>>>(outb, bnsum, bnsq, 256);
    int addRes = (l > 0 && (l % 2) == 0) ? 1 : 0;
    k_bn_norm<<<blocks(N * HID, TB), TB, 0, stream>>>(
        outb, bnsum, bnsq, bn_gamma + (size_t)l * HID, bn_beta + (size_t)l * HID,
        residual, h16, addRes, N * HID, 1.f / (float)N);
  }

  // --- pooling ---
  fill(cnt, B, 0.f);
  fill(psum, B * HID, 0.f);
  fill(patt, B * HID, 0.f);
  fill(pmax, B * HID, -INFINITY);
  fill(gmax, B, -INFINITY);
  fill(gsum, B, 0.f);
  k_pool_acc<<<blocks(N * HID, TB), TB, 0, stream>>>(h16, batch, cnt, psum, pmax, N * HID);
  k_gate<<<blocks(N, TB), TB, 0, stream>>>(h16, g1W, g1b, g2W, g2b, batch, gateb, gmax, (int)N);
  k_gate_exp<<<blocks(N, TB), TB, 0, stream>>>(batch, gmax, gateb, gsum, (int)N);
  k_att_pool<<<blocks(N * HID, TB), TB, 0, stream>>>(h16, batch, gateb, gsum, patt, N * HID);
  k_assemble<<<blocks(B * 384, TB), TB, 0, stream>>>(psum, pmax, patt, cnt, g16, B * 384);

  // --- MLP head (WMMA) ---
  {
    long waves = (B >> 4) * (256 >> 4);
    k_wmma_gemm<<<blocks(waves * 32, TB), TB, 0, stream>>>(
        g16, m1Wt, mlp1_b, nullptr, t1_16, (int)B, 384, 256, 1);
  }
  {
    long waves = (B >> 4) * (HID >> 4);
    k_wmma_gemm<<<blocks(waves * 32, TB), TB, 0, stream>>>(
        t1_16, m2Wt, mlp2_b, (float*)d_out, nullptr, (int)B, 256, HID, 1);
  }
}
